// MultiHeadAttentionEinsum_69879117906640
// MI455X (gfx1250) — matrix operand order repaired and checked by reading, not compiled, not run on hardware
//
#include <hip/hip_runtime.h>

// ---------------------------------------------------------------------------
// MI455X (gfx1250, wave32) multi-head causal attention.
//   K1: fused QKV projection GEMM (fp32 x, fp32 W -> bf16 Q/K/V in [B,H,N,64])
//   K2: flash attention per (qtile, head, batch) using v_wmma_f32_16x16x32_bf16
//       with global_load_async_to_lds_b128 staging of the K tile
//   K3: output projection GEMM + bias (bf16 ctx async-staged, fp32 W_out)
// ---------------------------------------------------------------------------

typedef __bf16 bf16_t;
typedef __attribute__((ext_vector_type(16))) __bf16 v16bf;
typedef __attribute__((ext_vector_type(8)))  float  v8f;

constexpr int  Bb   = 4;
constexpr int  Nn   = 2048;
constexpr int  Din  = 1024;
constexpr int  Dout = 1024;
constexpr int  Hh   = 16;
constexpr int  Dh   = 64;
constexpr int  Mrows = Bb * Nn;            // 8192 GEMM rows

// Pack two fp32 into two bf16 (native v_cvt path, RTE).
__device__ __forceinline__ unsigned packf2(float a, float b) {
  union { bf16_t h[2]; unsigned u; } t;
  t.h[0] = (bf16_t)a;
  t.h[1] = (bf16_t)b;
  return t.u;
}

union FragCast { v16bf v; uint4 q[2]; };
union U4BF { uint4 q; bf16_t h[8]; };

// Async copy of 16 bytes global -> LDS (per-lane), tracked by ASYNCcnt.
// GV addressing mode: 64-bit VGPR address, saddr = off.
__device__ __forceinline__ void async_copy_b128(void* lds_dst, const void* gsrc) {
  const unsigned lds_addr = (unsigned)(size_t)lds_dst;  // low 32b of generic
  asm volatile("global_load_async_to_lds_b128 %0, %1, off"
               :: "v"(lds_addr), "v"(gsrc)
               : "memory");
}
__device__ __forceinline__ void wait_async() {
  asm volatile("s_wait_asynccnt 0x0" ::: "memory");
}

// A-fragment (16x32 bf16, M x K): lane m=L%16, half=L/16 reads two 16B runs:
//   K[half*8 .. +8)   at byte offset half*16
//   K[16+half*8 ..)   at byte offset 32 + half*16
__device__ __forceinline__ v16bf load_a_frag(const bf16_t* base, int pitch) {
  const int lane = threadIdx.x & 31;
  const int m    = lane & 15;
  const int hf   = lane >> 4;
  const char* p  = (const char*)(base + m * pitch);
  FragCast f;
  f.q[0] = *(const uint4*)(p + hf * 16);
  f.q[1] = *(const uint4*)(p + 32 + hf * 16);
  return f.v;
}

// B-fragment (32x16 bf16, K x N) from a row-major [n][k] (i.e. B^T) tile:
// lane n=L%16, half=L/16 reads two 16B runs: K[half*8 .. +8), K[16+half*8 .. +8).
__device__ __forceinline__ v16bf load_b_frag(const bf16_t* base, int pitch) {
  const int lane = threadIdx.x & 31;
  const int n    = lane & 15;
  const int hf   = lane >> 4;
  const char* p  = (const char*)(base + n * pitch);
  FragCast f;
  f.q[0] = *(const uint4*)(p + hf * 16);
  f.q[1] = *(const uint4*)(p + 32 + hf * 16);
  return f.v;
}

__device__ __forceinline__ v8f wmma_bf16(v16bf a, v16bf b, v8f c) {
  return __builtin_amdgcn_wmma_f32_16x16x32_bf16(
      /*neg_a=*/false, a, /*neg_b=*/false, b,
      /*c_mod=*/(short)0, c, /*reuse_a=*/false, /*reuse_b=*/false);
}

__device__ __forceinline__ v8f v8f_zero() {
  v8f z;
#pragma unroll
  for (int i = 0; i < 8; ++i) z[i] = 0.0f;
  return z;
}

// ---------------------------------------------------------------------------
// Kernel 1: fused QKV projection.  Y[row, d] = sum_k x[row,k] * W[d,k]
// Tile: 128(M) x 64(N), K-step 32, 256 threads = 8 waves, wave = 16 rows.
// Output in bf16 head layout [B, H, N, 64].
// ---------------------------------------------------------------------------
__global__ __launch_bounds__(256) void qkv_proj_kernel(
    const float* __restrict__ x,
    const float* __restrict__ Wq, const float* __restrict__ Wk,
    const float* __restrict__ Wv,
    bf16_t* __restrict__ Qo, bf16_t* __restrict__ Ko, bf16_t* __restrict__ Vo) {
  __shared__ bf16_t As[128 * 40];   // pitch 40 bf16 (80B) -> conflict-free
  __shared__ bf16_t Bs[64 * 40];

  const int tid  = threadIdx.x;
  const int wid  = tid >> 5;
  const int lane = tid & 31;
  const int hf   = lane >> 4;
  const int nl   = lane & 15;
  const int m0   = blockIdx.x * 128;
  const int n0   = blockIdx.y * 64;

  const float* W   = (blockIdx.z == 0) ? Wq : ((blockIdx.z == 1) ? Wk : Wv);
  bf16_t*      Out = (blockIdx.z == 0) ? Qo : ((blockIdx.z == 1) ? Ko : Vo);

  v8f acc[4];
#pragma unroll
  for (int i = 0; i < 4; ++i) acc[i] = v8f_zero();

  for (int kt = 0; kt < Din / 32; ++kt) {
    const int k0 = kt * 32;
    // stage A (x): 128x32 fp32 -> bf16
#pragma unroll
    for (int i = 0; i < 4; ++i) {
      const int idx = tid + i * 256;
      const int row = idx >> 3, kq = idx & 7;
      float4 v = *(const float4*)(x + (size_t)(m0 + row) * Din + k0 + kq * 4);
      *(uint2*)(&As[row * 40 + kq * 4]) =
          make_uint2(packf2(v.x, v.y), packf2(v.z, v.w));
    }
    // stage B (W rows = output dims): 64x32 fp32 -> bf16
#pragma unroll
    for (int i = 0; i < 2; ++i) {
      const int idx = tid + i * 256;
      const int row = idx >> 3, kq = idx & 7;
      float4 v = *(const float4*)(W + (size_t)(n0 + row) * Din + k0 + kq * 4);
      *(uint2*)(&Bs[row * 40 + kq * 4]) =
          make_uint2(packf2(v.x, v.y), packf2(v.z, v.w));
    }
    __syncthreads();

    const v16bf af = load_a_frag(&As[wid * 16 * 40], 40);
#pragma unroll
    for (int ns = 0; ns < 4; ++ns) {
      const v16bf bfrag = load_b_frag(&Bs[ns * 16 * 40], 40);
      acc[ns] = wmma_bf16(af, bfrag, acc[ns]);
    }
    __syncthreads();
  }

  // epilogue: scatter to [B, H, N, 64] bf16
#pragma unroll
  for (int ns = 0; ns < 4; ++ns) {
#pragma unroll
    for (int r = 0; r < 8; ++r) {
      const int row = m0 + wid * 16 + r + hf * 8;
      const int col = n0 + ns * 16 + nl;
      const int bidx = row >> 11;          // row / 2048
      const int n    = row & 2047;
      const int h    = col >> 6;
      const int dh   = col & 63;
      const size_t o = ((((size_t)bidx * Hh + h) * Nn + n) * Dh + dh);
      Out[o] = (bf16_t)acc[ns][r];
    }
  }
}

// ---------------------------------------------------------------------------
// Kernel 2: flash attention.  One block = 64 query rows of one (b,h).
// 128 threads = 4 waves; wave handles 16 query rows (4 accum tiles of 16x16).
// ---------------------------------------------------------------------------
__global__ __launch_bounds__(128) void flash_attn_kernel(
    const bf16_t* __restrict__ Q, const bf16_t* __restrict__ K,
    const bf16_t* __restrict__ V, bf16_t* __restrict__ Ctx) {
  __shared__ bf16_t Qs[64 * 72];    // pitch 72 bf16 (144B)
  __shared__ bf16_t Ks[64 * 72];
  __shared__ bf16_t Vts[64 * 72];   // transposed: [dh][key]
  __shared__ bf16_t Ps[64 * 72];

  const int tid  = threadIdx.x;
  const int wid  = tid >> 5;                    // 0..3
  const int lane = tid & 31;
  const int hf   = lane >> 4;
  const int nl   = lane & 15;

  const int qt = blockIdx.x;
  const int h  = blockIdx.y;
  const int b  = blockIdx.z;
  const size_t headbase = ((size_t)b * Hh + h) * (size_t)Nn * Dh;
  const int q0 = qt * 64;

  // stage Q tile (scaled by 1/sqrt(64) = 0.125 folded in here)
#pragma unroll
  for (int i = 0; i < 4; ++i) {
    const int idx = tid + i * 128;
    const int row = idx >> 3, ch = idx & 7;
    U4BF t, o;
    t.q = *(const uint4*)(Q + headbase + (size_t)(q0 + row) * Dh + ch * 8);
#pragma unroll
    for (int e = 0; e < 8; ++e) o.h[e] = (bf16_t)((float)t.h[e] * 0.125f);
    *(uint4*)(&Qs[row * 72 + ch * 8]) = o.q;
  }
  __syncthreads();

  v16bf qf[2];
  qf[0] = load_a_frag(&Qs[wid * 16 * 72], 72);        // K(dh) 0..31
  qf[1] = load_a_frag(&Qs[wid * 16 * 72 + 32], 72);   // K(dh) 32..63

  float mrow[8], lrow[8];
  v8f o[4];
#pragma unroll
  for (int r = 0; r < 8; ++r) { mrow[r] = -1.0e30f; lrow[r] = 0.0f; }
#pragma unroll
  for (int t = 0; t < 4; ++t) o[t] = v8f_zero();

  for (int j = 0; j <= qt; ++j) {
    __syncthreads();  // previous Ks/Vts consumers done
    const size_t kb = headbase + (size_t)j * 64 * Dh;

    // prefetch next K/V tiles into cache (global_prefetch_b8)
    if (j < qt) {
      const size_t kb2 = kb + (size_t)64 * Dh;
      __builtin_prefetch(K + kb2 + (size_t)tid * 32, 0, 1);
      __builtin_prefetch(V + kb2 + (size_t)tid * 32, 0, 1);
    }

    // K tile: pure byte copy -> use async DMA to LDS (ASYNCcnt)
#pragma unroll
    for (int i = 0; i < 4; ++i) {
      const int idx = tid + i * 128;
      const int row = idx >> 3, ch = idx & 7;
      async_copy_b128(&Ks[row * 72 + ch * 8],
                      K + kb + (size_t)row * Dh + ch * 8);
    }
    // V tile: store transposed [dh][key] (needs VGPR pass)
#pragma unroll
    for (int i = 0; i < 4; ++i) {
      const int idx = tid + i * 128;
      const int key = idx >> 3, ch = idx & 7;
      U4BF t;
      t.q = *(const uint4*)(V + kb + (size_t)key * Dh + ch * 8);
#pragma unroll
      for (int e = 0; e < 8; ++e) Vts[(ch * 8 + e) * 72 + key] = t.h[e];
    }
    wait_async();
    __syncthreads();

    // S = Q * K^T  (16 rows x 64 keys per wave)
    v8f s[4];
#pragma unroll
    for (int ns = 0; ns < 4; ++ns) {
      s[ns] = v8f_zero();
#pragma unroll
      for (int kc = 0; kc < 2; ++kc) {
        const v16bf bfrag = load_b_frag(&Ks[ns * 16 * 72 + kc * 32], 72);
        s[ns] = wmma_bf16(qf[kc], bfrag, s[ns]);
      }
    }

    // causal mask on the diagonal tile
    if (j == qt) {
#pragma unroll
      for (int ns = 0; ns < 4; ++ns) {
        const int kg = j * 64 + ns * 16 + nl;
#pragma unroll
        for (int r = 0; r < 8; ++r) {
          const int qg = q0 + wid * 16 + r + hf * 8;
          if (kg > qg) s[ns][r] = -1.0e30f;
        }
      }
    }

    // online softmax: row stats per (r, half) — butterfly within 16 lanes
    float alpha[8];
#pragma unroll
    for (int r = 0; r < 8; ++r) {
      float v = fmaxf(fmaxf(s[0][r], s[1][r]), fmaxf(s[2][r], s[3][r]));
#pragma unroll
      for (int msk = 1; msk < 16; msk <<= 1) v = fmaxf(v, __shfl_xor(v, msk, 32));
      const float mnew = fmaxf(mrow[r], v);
      alpha[r] = __expf(mrow[r] - mnew);
      mrow[r] = mnew;
    }
#pragma unroll
    for (int ns = 0; ns < 4; ++ns)
#pragma unroll
      for (int r = 0; r < 8; ++r) s[ns][r] = __expf(s[ns][r] - mrow[r]);
#pragma unroll
    for (int r = 0; r < 8; ++r) {
      float v = s[0][r] + s[1][r] + s[2][r] + s[3][r];
#pragma unroll
      for (int msk = 1; msk < 16; msk <<= 1) v += __shfl_xor(v, msk, 32);
      lrow[r] = lrow[r] * alpha[r] + v;
    }
#pragma unroll
    for (int t = 0; t < 4; ++t)
#pragma unroll
      for (int r = 0; r < 8; ++r) o[t][r] *= alpha[r];

    // restage P (bf16) into wave-private LDS rows, then O += P * V
#pragma unroll
    for (int ns = 0; ns < 4; ++ns)
#pragma unroll
      for (int r = 0; r < 8; ++r) {
        const int qr = wid * 16 + r + hf * 8;
        Ps[qr * 72 + ns * 16 + nl] = (bf16_t)s[ns][r];
      }
    asm volatile("s_wait_dscnt 0x0" ::: "memory");  // wave-local LDS RAW fence

    v16bf pf[2];
    pf[0] = load_a_frag(&Ps[wid * 16 * 72], 72);
    pf[1] = load_a_frag(&Ps[wid * 16 * 72 + 32], 72);
#pragma unroll
    for (int t = 0; t < 4; ++t)
#pragma unroll
      for (int kc = 0; kc < 2; ++kc) {
        const v16bf vf = load_b_frag(&Vts[t * 16 * 72 + kc * 32], 72);
        o[t] = wmma_bf16(pf[kc], vf, o[t]);
      }
  }

  // epilogue: ctx[b, n, h*64 + dh] in bf16
#pragma unroll
  for (int r = 0; r < 8; ++r) {
    const float inv = 1.0f / lrow[r];
    const int nrow = q0 + wid * 16 + r + hf * 8;
#pragma unroll
    for (int t = 0; t < 4; ++t) {
      const int d = h * 64 + t * 16 + nl;
      const size_t oidx = ((size_t)b * Nn + nrow) * Dout + d;
      Ctx[oidx] = (bf16_t)(o[t][r] * inv);
    }
  }
}

// ---------------------------------------------------------------------------
// Kernel 3: out = ctx @ W_out^T + b_out   (bf16 A, fp32->bf16 B, fp32 out)
// ---------------------------------------------------------------------------
__global__ __launch_bounds__(256) void out_proj_kernel(
    const bf16_t* __restrict__ Ctx, const float* __restrict__ Wo,
    const float* __restrict__ bias, float* __restrict__ out) {
  __shared__ bf16_t As[128 * 40];
  __shared__ bf16_t Bs[64 * 40];

  const int tid  = threadIdx.x;
  const int wid  = tid >> 5;
  const int lane = tid & 31;
  const int hf   = lane >> 4;
  const int nl   = lane & 15;
  const int m0   = blockIdx.x * 128;
  const int n0   = blockIdx.y * 64;

  v8f acc[4];
#pragma unroll
  for (int i = 0; i < 4; ++i) acc[i] = v8f_zero();

  for (int kt = 0; kt < Dout / 32; ++kt) {
    const int k0 = kt * 32;
    // stage A (ctx, already bf16): async DMA copy 128x32
#pragma unroll
    for (int i = 0; i < 2; ++i) {
      const int idx = tid + i * 256;
      const int row = idx >> 2, ch = idx & 3;
      async_copy_b128(&As[row * 40 + ch * 8],
                      Ctx + (size_t)(m0 + row) * Dout + k0 + ch * 8);
    }
    // stage B (W_out rows): 64x32 fp32 -> bf16
#pragma unroll
    for (int i = 0; i < 2; ++i) {
      const int idx = tid + i * 256;
      const int row = idx >> 3, kq = idx & 7;
      float4 v = *(const float4*)(Wo + (size_t)(n0 + row) * Dout + k0 + kq * 4);
      *(uint2*)(&Bs[row * 40 + kq * 4]) =
          make_uint2(packf2(v.x, v.y), packf2(v.z, v.w));
    }
    wait_async();
    __syncthreads();

    const v16bf af = load_a_frag(&As[wid * 16 * 40], 40);
#pragma unroll
    for (int ns = 0; ns < 4; ++ns) {
      const v16bf bfrag = load_b_frag(&Bs[ns * 16 * 40], 40);
      acc[ns] = wmma_bf16(af, bfrag, acc[ns]);
    }
    __syncthreads();
  }

#pragma unroll
  for (int ns = 0; ns < 4; ++ns) {
    const int col = n0 + ns * 16 + nl;
    const float bv = bias[col];
#pragma unroll
    for (int r = 0; r < 8; ++r) {
      const int row = m0 + wid * 16 + r + hf * 8;
      out[(size_t)row * Dout + col] = acc[ns][r] + bv;
    }
  }
}

// ---------------------------------------------------------------------------
// Host-side launcher.
// d_in order: x, W_q, W_k, W_v, W_out, b_out (all fp32).
// Workspace: Q,K,V ([B,H,N,64] bf16, 16 MB each) + ctx ([B,N,1024] bf16, 16 MB).
// ---------------------------------------------------------------------------
extern "C" void kernel_launch(void* const* d_in, const int* in_sizes, int n_in,
                              void* d_out, int out_size, void* d_ws, size_t ws_size,
                              hipStream_t stream) {
  const float* x  = (const float*)d_in[0];
  const float* Wq = (const float*)d_in[1];
  const float* Wk = (const float*)d_in[2];
  const float* Wv = (const float*)d_in[3];
  const float* Wo = (const float*)d_in[4];
  const float* bo = (const float*)d_in[5];
  float* out = (float*)d_out;

  const size_t qkv_elems = (size_t)Bb * Hh * Nn * Dh;   // 8,388,608
  bf16_t* Q   = (bf16_t*)d_ws;
  bf16_t* K   = Q + qkv_elems;
  bf16_t* V   = K + qkv_elems;
  bf16_t* Ctx = V + qkv_elems;

  qkv_proj_kernel<<<dim3(Mrows / 128, Dout / 64, 3), 256, 0, stream>>>(
      x, Wq, Wk, Wv, Q, K, V);
  flash_attn_kernel<<<dim3(Nn / 64, Hh, Bb), 128, 0, stream>>>(Q, K, V, Ctx);
  out_proj_kernel<<<dim3(Mrows / 128, Dout / 64), 256, 0, stream>>>(
      Ctx, Wo, bo, out);
}
